// AdaptiveDiffusionConv_15822659518585
// MI455X (gfx1250) — matrix operand order repaired and checked by reading, not compiled, not run on hardware
//
#include <hip/hip_runtime.h>
#include <hip/hip_bf16.h>
#include <stddef.h>

typedef __bf16 bf16_t;
typedef __attribute__((ext_vector_type(16))) __bf16 v16bf;
typedef __attribute__((ext_vector_type(8)))  __bf16 v8bf;
typedef __attribute__((ext_vector_type(8)))  float  v8f;

#define NN 1024
#define BB 16
#define FF 16
#define TT 12
#define CC (FF * TT)   // 192 fused (f,t) columns

// ---------------------------------------------------------------------------
// Kernel 1: At[b][n][m] = bf16( adj[m][n] * att[b][m][n] )   (transposed a)
// LDS-tiled 32x32 transpose so both read and write are coalesced.
// ---------------------------------------------------------------------------
__global__ __launch_bounds__(256) void k_make_at(const float* __restrict__ adj,
                                                 const float* __restrict__ att,
                                                 bf16_t* __restrict__ At) {
  __shared__ bf16_t tile[32][33];
  const int b  = blockIdx.z;
  const int m0 = blockIdx.y * 32;
  const int n0 = blockIdx.x * 32;
  const float* attb = att + (size_t)b * NN * NN;
#pragma unroll
  for (int r = threadIdx.y; r < 32; r += 8) {
    const size_t off = (size_t)(m0 + r) * NN + (n0 + threadIdx.x);
    tile[r][threadIdx.x] = (bf16_t)(adj[off] * attb[off]);
  }
  __syncthreads();
  bf16_t* Atb = At + (size_t)b * NN * NN;
#pragma unroll
  for (int r = threadIdx.y; r < 32; r += 8) {
    Atb[(size_t)(n0 + r) * NN + (m0 + threadIdx.x)] = tile[threadIdx.x][r];
  }
}

// ---------------------------------------------------------------------------
// Kernel 2: Xt[b][c][m] = bf16( x[b][m][c] ), c = f*T + t  (K-major X)
// ---------------------------------------------------------------------------
__global__ __launch_bounds__(256) void k_make_xt(const float* __restrict__ x,
                                                 bf16_t* __restrict__ Xt) {
  __shared__ bf16_t tile[32][33];
  const int b  = blockIdx.z;
  const int m0 = blockIdx.y * 32;   // row in x (node index)
  const int c0 = blockIdx.x * 32;   // column (f*T+t)
  const float* xb = x + (size_t)b * NN * CC;
#pragma unroll
  for (int r = threadIdx.y; r < 32; r += 8) {
    tile[r][threadIdx.x] = (bf16_t)xb[(size_t)(m0 + r) * CC + (c0 + threadIdx.x)];
  }
  __syncthreads();
  bf16_t* Xtb = Xt + (size_t)b * CC * NN;
#pragma unroll
  for (int r = threadIdx.y; r < 32; r += 8) {
    Xtb[(size_t)(c0 + r) * NN + (m0 + threadIdx.x)] = tile[threadIdx.x][r];
  }
}

// ---------------------------------------------------------------------------
// Kernel 3: batched GEMM  R(n,c) = sum_m At[n][m] * Bt[c][m]   (i.e. a^T @ B)
//   At : [B][1024][1024] bf16 (row n, K along m)
//   Bt : [B][ 192][1024] bf16 (row c, K along m)
//   Rt : [B][ 192][1024] bf16 (row c, col n)  -- transposed result for chaining
//
// 6 waves / block (192 thr); wave = 32 rows x 64 cols (2 M-tiles x 4 N-tiles,
// 8 accumulators); block = 64 rows x 192 cols. Two-stage register double
// buffer over K so next-step loads overlap current-step WMMAs (avoids the
// s_wait_loadcnt 0 full drains seen with a single fragment buffer).
//
// WMMA fragment layouts (A/B: contiguous K order chosen by this code; C/D as in CDNA5 ISA 7.12.2):
//   A 16x32 bf16: lane<16 holds M=lane, K {0..15};       lane>=16 K {16..31}
//   B 32x16 bf16: lane<16 holds N=lane, K {0..15};       lane>=16 K {16..31}
//   C/D 16x16 f32: VGPR i = row (lane<16 ? i : 8+i), col = lane&15
// ---------------------------------------------------------------------------
struct GemmFrag {
  v16bf a0, a1;   // two 16-row A tiles (rows row0..+15, row0+16..+31)
  v16bf b[4];     // four 16-col B tiles
};

__device__ __forceinline__ v16bf load_a_frag(const bf16_t* arow, int kb) {
  v8bf lo = *(const v8bf*)(arow + kb);
  v8bf hi = *(const v8bf*)(arow + kb + 8);
  v16bf f;
#pragma unroll
  for (int i = 0; i < 8; ++i) { f[i] = lo[i]; f[i + 8] = hi[i]; }
  return f;
}

__device__ __forceinline__ void load_frag(GemmFrag& f,
                                          const bf16_t* arow0, const bf16_t* arow1,
                                          const bf16_t* bbase, int kb) {
  __builtin_prefetch(arow0 + kb + 128, 0, 1);   // -> global_prefetch_b8
  f.a0 = load_a_frag(arow0, kb);
  f.a1 = load_a_frag(arow1, kb);
#pragma unroll
  for (int ct = 0; ct < 4; ++ct)
    f.b[ct] = *(const v16bf*)(bbase + (size_t)ct * 16 * NN + kb);
}

__device__ __forceinline__ void compute_frag(const GemmFrag& f, v8f acc[8]) {
#pragma unroll
  for (int ct = 0; ct < 4; ++ct) {
    acc[ct] = __builtin_amdgcn_wmma_f32_16x16x32_bf16(
        false, f.a0, false, f.b[ct], (short)0, acc[ct], false, false);
    acc[4 + ct] = __builtin_amdgcn_wmma_f32_16x16x32_bf16(
        false, f.a1, false, f.b[ct], (short)0, acc[4 + ct], false, false);
  }
}

__global__ __launch_bounds__(192) void k_gemm_atb(const bf16_t* __restrict__ At,
                                                  const bf16_t* __restrict__ Bt,
                                                  bf16_t* __restrict__ Rt) {
  const int lane = threadIdx.x & 31;
  const int wave = threadIdx.x >> 5;          // 0..5
  const int b    = blockIdx.y;
  const int rg   = wave & 1;                  // row group   (2 x 32 rows)
  const int cg   = wave >> 1;                 // col group   (3 x 64 cols)
  const int row0 = blockIdx.x * 64 + rg * 32;
  const int col0 = cg * 64;
  const int lrow = lane & 15;
  const int hi   = lane >> 4;

  const bf16_t* Ab = At + (size_t)b * NN * NN;
  const bf16_t* Bb = Bt + (size_t)b * CC * NN;

  // Per-lane base pointers matching the fragment layouts above.
  const bf16_t* arow0 = Ab + (size_t)(row0 + lrow) * NN + hi * 16;
  const bf16_t* arow1 = arow0 + (size_t)16 * NN;
  const bf16_t* bbase = Bb + (size_t)(col0 + lrow) * NN + hi * 16;

  v8f acc[8];
#pragma unroll
  for (int i = 0; i < 8; ++i) acc[i] = (v8f){0.f, 0.f, 0.f, 0.f, 0.f, 0.f, 0.f, 0.f};

  // Two-stage software pipeline over K (NN/32 = 32 steps, even count).
  GemmFrag fa, fb;
  load_frag(fa, arow0, arow1, bbase, 0);
#pragma unroll 1
  for (int kb = 0; kb < NN - 64; kb += 64) {
    load_frag(fb, arow0, arow1, bbase, kb + 32);
    compute_frag(fa, acc);
    load_frag(fa, arow0, arow1, bbase, kb + 64);
    compute_frag(fb, acc);
  }
  load_frag(fb, arow0, arow1, bbase, NN - 32);
  compute_frag(fa, acc);
  compute_frag(fb, acc);

  // Store transposed result: lane writes 8 consecutive n's of one column c.
  bf16_t* Rb = Rt + (size_t)b * CC * NN;
#pragma unroll
  for (int mt = 0; mt < 2; ++mt) {
    const int rbase = row0 + mt * 16 + hi * 8;
#pragma unroll
    for (int ct = 0; ct < 4; ++ct) {
      bf16_t* dst = Rb + (size_t)(col0 + ct * 16 + lrow) * NN + rbase;
      v8bf o;
#pragma unroll
      for (int i = 0; i < 8; ++i) o[i] = (bf16_t)acc[mt * 4 + ct][i];
      *(v8bf*)dst = o;
    }
  }
}

// ---------------------------------------------------------------------------
// Kernel 4: out[b,n,o,t] = relu( sum_f x[b,n,f,t]*Th0[f,o]
//                               + R1[n,f,t]*Th1[f,o] + R2[n,f,t]*Th2[f,o] )
// One thread per (b,n,t); 16 output channels in registers; Theta in LDS.
// ---------------------------------------------------------------------------
__global__ __launch_bounds__(256) void k_project(const float* __restrict__ x,
                                                 const bf16_t* __restrict__ R1t,
                                                 const bf16_t* __restrict__ R2t,
                                                 const float* __restrict__ Theta,
                                                 float* __restrict__ out) {
  __shared__ float th[3 * FF * 16];
  for (int i = threadIdx.x; i < 3 * FF * 16; i += 256) th[i] = Theta[i];
  __syncthreads();

  const int idx = blockIdx.x * 256 + threadIdx.x;   // over B*N*T = 196608
  const int t  = idx % TT;
  const int bn = idx / TT;                           // b*N + n
  const int n  = bn & (NN - 1);
  const int b  = bn >> 10;

  const float*  xp = x + (size_t)bn * CC + t;                         // stride T per f
  const bf16_t* r1 = R1t + (size_t)b * CC * NN + (size_t)t * NN + n;  // stride T*N per f
  const bf16_t* r2 = R2t + (size_t)b * CC * NN + (size_t)t * NN + n;

  float acc[16];
#pragma unroll
  for (int o = 0; o < 16; ++o) acc[o] = 0.f;

#pragma unroll
  for (int f = 0; f < FF; ++f) {
    const float xv = xp[f * TT];
    const float v1 = (float)r1[(size_t)f * TT * NN];
    const float v2 = (float)r2[(size_t)f * TT * NN];
    const float* t0 = th + f * 16;
    const float* t1 = th + 256 + f * 16;
    const float* t2 = th + 512 + f * 16;
#pragma unroll
    for (int o = 0; o < 16; ++o)
      acc[o] = fmaf(xv, t0[o], fmaf(v1, t1[o], fmaf(v2, t2[o], acc[o])));
  }

  float* op = out + (size_t)bn * CC + t;   // [B][N][16][12]
#pragma unroll
  for (int o = 0; o < 16; ++o) op[o * TT] = fmaxf(acc[o], 0.f);
}

// ---------------------------------------------------------------------------
extern "C" void kernel_launch(void* const* d_in, const int* in_sizes, int n_in,
                              void* d_out, int out_size, void* d_ws, size_t ws_size,
                              hipStream_t stream) {
  const float* x     = (const float*)d_in[0];   // [16,1024,16,12]
  const float* att   = (const float*)d_in[1];   // [16,1024,1024]
  const float* adj   = (const float*)d_in[2];   // [1024,1024]
  const float* Theta = (const float*)d_in[3];   // [3,16,16]
  float* out = (float*)d_out;

  char* ws = (char*)d_ws;
  const size_t sz_At = (size_t)BB * NN * NN * sizeof(bf16_t);   // 32 MB
  const size_t sz_Xt = (size_t)BB * CC * NN * sizeof(bf16_t);   //  6 MB
  bf16_t* At  = (bf16_t*)(ws);
  bf16_t* Xt  = (bf16_t*)(ws + sz_At);
  bf16_t* R1t = (bf16_t*)(ws + sz_At + sz_Xt);
  bf16_t* R2t = (bf16_t*)(ws + sz_At + 2 * sz_Xt);

  // 1) a^T in bf16 (fused elementwise multiply + transpose)
  k_make_at<<<dim3(NN / 32, NN / 32, BB), dim3(32, 8), 0, stream>>>(adj, att, At);
  // 2) X^T in bf16
  k_make_xt<<<dim3(CC / 32, NN / 32, BB), dim3(32, 8), 0, stream>>>(x, Xt);
  // 3) R1 = a^T @ X        (WMMA bf16)
  k_gemm_atb<<<dim3(NN / 64, BB), 192, 0, stream>>>(At, Xt, R1t);
  // 4) R2 = a^T @ R1  ( == (a@a)^T @ X, avoids the N^3 matmul entirely )
  k_gemm_atb<<<dim3(NN / 64, BB), 192, 0, stream>>>(At, R1t, R2t);
  // 5) fused Theta projection + ReLU
  k_project<<<dim3((BB * NN * TT) / 256), 256, 0, stream>>>(x, R1t, R2t, Theta, out);
}
